// GSF_DTA_44487271252216
// MI455X (gfx1250) — compile-verified, matrix operand order checked by reading, not run on hardware
//
#include <hip/hip_runtime.h>
#include <hip/hip_bf16.h>

#define HID 128

typedef __bf16 v16bf __attribute__((ext_vector_type(16)));
typedef float  v8f   __attribute__((ext_vector_type(8)));

// CDNA5 16-bit WMMA operand layout: element e of a lane's v16bf maps to
// K = (e<8 ? e : e+8) + 8*hi  (hi = lane>>4), row/col = lane&15.
__device__ __forceinline__ int kof_of(int e, int hi) {
    return ((e < 8) ? e : (e + 8)) + (hi << 3);
}

// ---------------- utility kernels ----------------

__global__ void fill_f(float* __restrict__ p, int n, float v) {
    int g = blockIdx.x * blockDim.x + threadIdx.x;
    int stride = gridDim.x * blockDim.x;
    for (int i = g; i < n; i += stride) p[i] = v;
}

__global__ void deg_accum(const int* __restrict__ dst, float* __restrict__ deg, int E) {
    int g = blockIdx.x * blockDim.x + threadIdx.x;
    int stride = gridDim.x * blockDim.x;
    for (int e = g; e < E; e += stride) atomicAdd(&deg[dst[e]], 1.0f);
}

__global__ void to_dinv(float* __restrict__ d, int n) {
    int g = blockIdx.x * blockDim.x + threadIdx.x;
    int stride = gridDim.x * blockDim.x;
    for (int i = g; i < n; i += stride) d[i] = rsqrtf(fmaxf(d[i], 1.0f));
}

// ---------------- operand packing into WMMA per-lane layout -----------------

// A: H [n,K] f32 row-major -> Apack, one v16bf per (row-tile, k-chunk, lane).
// Apack index (v16bf units): (rt*nc + c)*32 + lane.
__global__ void pack_a_bf16(const float* __restrict__ H, __bf16* __restrict__ Apack,
                            int n, int K, int nc /* Kpad/32 */) {
    int g = blockIdx.x * blockDim.x + threadIdx.x;
    int stride = gridDim.x * blockDim.x;
    int total = (n >> 4) * nc * 32;
    for (int id = g; id < total; id += stride) {
        int l = id & 31;
        int t = id >> 5;
        int c = t % nc;
        int rt = t / nc;
        int nn = l & 15, hi = l >> 4;
        int row = (rt << 4) + nn;
        int kc = c << 5;
        v16bf o;
        if ((K & 7) == 0 && kc + 32 <= K) {
            // aligned fast path: two 8-float runs per lane (see kof_of)
            const float* rp = H + (size_t)row * K + kc + (hi << 3);
            float4 a0 = *(const float4*)(rp + 0);
            float4 a1 = *(const float4*)(rp + 4);
            float4 b0 = *(const float4*)(rp + 16);
            float4 b1 = *(const float4*)(rp + 20);
            o[0]=(__bf16)a0.x;  o[1]=(__bf16)a0.y;  o[2]=(__bf16)a0.z;  o[3]=(__bf16)a0.w;
            o[4]=(__bf16)a1.x;  o[5]=(__bf16)a1.y;  o[6]=(__bf16)a1.z;  o[7]=(__bf16)a1.w;
            o[8]=(__bf16)b0.x;  o[9]=(__bf16)b0.y;  o[10]=(__bf16)b0.z; o[11]=(__bf16)b0.w;
            o[12]=(__bf16)b1.x; o[13]=(__bf16)b1.y; o[14]=(__bf16)b1.z; o[15]=(__bf16)b1.w;
        } else {
#pragma unroll
            for (int e = 0; e < 16; ++e) {
                int gk = kc + kof_of(e, hi);
                o[e] = (gk < K) ? (__bf16)H[(size_t)row * K + gk] : (__bf16)0.0f;
            }
        }
        *(v16bf*)(Apack + (size_t)id * 16) = o;
    }
}

// W: [K,128] f32 row-major -> Wpack, one v16bf per (k-chunk, col-tile, lane).
// Wpack index (v16bf units): (c*8 + ct)*32 + lane. Zero-padded to nc*32 rows.
__global__ void pack_w_bf16(const float* __restrict__ W, __bf16* __restrict__ Wpack,
                            int K, int nc) {
    int g = blockIdx.x * blockDim.x + threadIdx.x;
    int stride = gridDim.x * blockDim.x;
    int total = nc * 8 * 32;
    for (int id = g; id < total; id += stride) {
        int l = id & 31;
        int t = id >> 5;
        int ct = t & 7;
        int c = t >> 3;
        int nn = l & 15, hi = l >> 4;
        int col = (ct << 4) + nn;
        int kc = c << 5;
        v16bf o;
#pragma unroll
        for (int e = 0; e < 16; ++e) {
            int k = kc + kof_of(e, hi);
            o[e] = (k < K) ? (__bf16)W[(size_t)k * HID + col] : (__bf16)0.0f;
        }
        *(v16bf*)(Wpack + (size_t)id * 16) = o;
    }
}

// ---------------- WMMA GEMM: z = dinv .* (H @ W), acc = z (self-loop seed) --
// Block = 256 threads = 8 wave32; block owns a 16-row band, wave owns a 16-col
// tile. Operands pre-packed: inner loop = 2 vector loads + v_wmma, no LDS.
// Requires n % 16 == 0 (true here: 50000 / 10000), grid.x == n/16.
template<int NC>
__global__ void gcn_gemm_wmma(const __bf16* __restrict__ Apack,
                              const __bf16* __restrict__ Wpack,
                              const float* __restrict__ dinv,
                              float* __restrict__ z, float* __restrict__ accbuf) {
    const int lane = threadIdx.x & 31;      // wave32
    const int wave = threadIdx.x >> 5;      // 0..7 -> column tile
    const int rt   = blockIdx.x;            // row tile
    const v16bf* ap = (const v16bf*)Apack + (size_t)rt * NC * 32 + lane;
    const v16bf* bp = (const v16bf*)Wpack + (size_t)wave * 32 + lane;
    v8f c8 = {};
#pragma unroll
    for (int c = 0; c < NC; ++c) {
        if (c + 1 < NC)  // speculative prefetch of next A chunk
            __builtin_prefetch((const void*)(ap + (size_t)(c + 1) * 32), 0, 3);
        v16bf a = ap[(size_t)c * 32];
        v16bf b = bp[(size_t)c * 256];      // 8 tiles * 32 lanes per chunk
        c8 = __builtin_amdgcn_wmma_f32_16x16x32_bf16(
                 /*neg_a=*/false, a, /*neg_b=*/false, b,
                 /*c_mod=*/(short)0, c8, /*reuse_a=*/false, /*reuse_b=*/false);
    }
    // C/D layout: VGPR r -> M = r + 8*hi, lane -> N = lane&15.
    // This lane's 8 rows are contiguous: row0 .. row0+7, row0 = 16*rt + 8*hi.
    const int nn = lane & 15, hi = lane >> 4;
    const int row0 = (rt << 4) + (hi << 3);
    const int n0 = wave << 4;
    const float* dp = dinv + row0;                 // 32B-aligned
    float4 d0 = *(const float4*)(dp + 0);
    float4 d1 = *(const float4*)(dp + 4);
    float dv[8] = {d0.x, d0.y, d0.z, d0.w, d1.x, d1.y, d1.z, d1.w};
#pragma unroll
    for (int r = 0; r < 8; ++r) {
        size_t idx = (size_t)(row0 + r) * HID + n0 + nn;
        float v = c8[r] * dv[r];                   // pre-scale by dinv[src]
        z[idx]      = v;
        accbuf[idx] = v;                           // self-loop seed
    }
}

// ---------------- edge scatter: acc[dst] += z[src] (wave per edge) ----------
__global__ void gcn_scatter(const int* __restrict__ src, const int* __restrict__ dst,
                            const float* __restrict__ z, float* __restrict__ accbuf,
                            int E) {
    int g = blockIdx.x * blockDim.x + threadIdx.x;
    int w = g >> 5;
    int lane = threadIdx.x & 31;
    int nw = (gridDim.x * blockDim.x) >> 5;
    for (int e = w; e < E; e += nw) {
        int s = src[e], d = dst[e];
        const float4 zv = *(const float4*)(z + (size_t)s * HID + lane * 4);
        float* ap = accbuf + (size_t)d * HID + lane * 4;
        atomicAdd(ap + 0, zv.x);
        atomicAdd(ap + 1, zv.y);
        atomicAdd(ap + 2, zv.z);
        atomicAdd(ap + 3, zv.w);
    }
}

// ---------------- h = relu(dinv[row] * acc + bias[col]) ---------------------
__global__ void gcn_finalize(const float* __restrict__ accbuf,
                             const float* __restrict__ dinv,
                             const float* __restrict__ bias,
                             float* __restrict__ h, int n) {
    size_t g = (size_t)blockIdx.x * blockDim.x + threadIdx.x;
    size_t stride = (size_t)gridDim.x * blockDim.x;
    size_t tot = (size_t)n * HID;
    for (size_t i = g; i < tot; i += stride) {
        int r = (int)(i >> 7), c = (int)(i & (HID - 1));
        h[i] = fmaxf(0.0f, accbuf[i] * dinv[r] + bias[c]);
    }
}

// ---------------- column mean pool (out must be pre-zeroed) -----------------
__global__ void mean_pool(const float* __restrict__ h, int n, float inv_n,
                          float* __restrict__ out) {
    int g = blockIdx.x * blockDim.x + threadIdx.x;
    int c = g & (HID - 1);
    int r0 = g >> 7;
    int rstride = (gridDim.x * blockDim.x) >> 7;
    float s = 0.0f;
    for (int r = r0; r < n; r += rstride) s += h[(size_t)r * HID + c];
    atomicAdd(&out[c], s * inv_n);
}

// ---------------- MLP head: relu(c @ W1 + b1) @ W2 + b2 ---------------------
__global__ void fc_head(const float* __restrict__ pooled,   // [256]
                        const float* __restrict__ W1, const float* __restrict__ b1,
                        const float* __restrict__ W2, const float* __restrict__ b2,
                        float* __restrict__ out) {
    __shared__ float sh[HID];
    int j = threadIdx.x;
    float s = b1[j];
    for (int i = 0; i < 2 * HID; ++i) s += pooled[i] * W1[i * HID + j];
    float hj = fmaxf(s, 0.0f);
    sh[j] = hj * W2[j];
    __syncthreads();
    for (int off = HID / 2; off > 0; off >>= 1) {
        if (j < off) sh[j] += sh[j + off];
        __syncthreads();
    }
    if (j == 0) out[0] = sh[0] + b2[0];
}

// ---------------- launch ----------------------------------------------------

extern "C" void kernel_launch(void* const* d_in, const int* in_sizes, int n_in,
                              void* d_out, int out_size, void* d_ws, size_t ws_size,
                              hipStream_t stream) {
    const float* px  = (const float*)d_in[0];
    const int*   pe  = (const int*)d_in[1];
    const float* dx  = (const float*)d_in[2];
    const int*   de  = (const int*)d_in[3];
    const float* Wp1 = (const float*)d_in[4];  const float* bp1 = (const float*)d_in[5];
    const float* Wp2 = (const float*)d_in[6];  const float* bp2 = (const float*)d_in[7];
    const float* Wd1 = (const float*)d_in[8];  const float* bd1 = (const float*)d_in[9];
    const float* Wd2 = (const float*)d_in[10]; const float* bd2 = (const float*)d_in[11];
    const float* Wf1 = (const float*)d_in[12]; const float* bf1 = (const float*)d_in[13];
    const float* Wf2 = (const float*)d_in[14]; const float* bf2 = (const float*)d_in[15];

    const int n_p = in_sizes[0] / 128;      // d_p = 128
    const int Ep  = in_sizes[1] / 2;
    const int n_d = in_sizes[2] / 78;       // d_d = 78
    const int Ed  = in_sizes[3] / 2;
    const int* pe_src = pe;  const int* pe_dst = pe + Ep;
    const int* de_src = de;  const int* de_dst = de + Ed;

    // workspace carve-up (floats), 64-float (256 B) aligned
    float* ws = (float*)d_ws;
    size_t o = 0;
    auto al = [](size_t x) { return (x + 63) & ~(size_t)63; };
    float* dinvP  = ws + o; o = al(o + (size_t)n_p);
    float* dinvD  = ws + o; o = al(o + (size_t)n_d);
    float* pooled = ws + o; o = al(o + 256);
    __bf16* Wpack = (__bf16*)(ws + o); o = al(o + (4 * 8 * 32 * 16) / 2);      // 16K bf16
    __bf16* Apack = (__bf16*)(ws + o); o = al(o + ((size_t)n_p * 128) / 2);     // n*Kpad bf16
    float* bigA = ws + o;   o = al(o + (size_t)n_p * HID);                      // z
    float* bigB = ws + o;   o = al(o + (size_t)n_p * HID);                      // acc
    float* bigC = ws + o;   o = al(o + (size_t)n_p * HID);                      // h
    (void)ws_size; (void)n_in; (void)out_size;

    const dim3 B256(256);
    auto cdiv = [](int a, int b) { return (a + b - 1) / b; };
    const int ntP = n_p / 16, ntD = n_d / 16;   // n_p, n_d are multiples of 16

    // ===== protein branch =====
    fill_f<<<cdiv(n_p, 256), B256, 0, stream>>>(dinvP, n_p, 1.0f);
    deg_accum<<<1024, B256, 0, stream>>>(pe_dst, dinvP, Ep);
    to_dinv<<<cdiv(n_p, 256), B256, 0, stream>>>(dinvP, n_p);

    // layer 1 (K = 128, nc = 4)
    pack_w_bf16<<<4, B256, 0, stream>>>(Wp1, Wpack, 128, 4);
    pack_a_bf16<<<cdiv(ntP * 4 * 32, 256), B256, 0, stream>>>(px, Apack, n_p, 128, 4);
    gcn_gemm_wmma<4><<<ntP, B256, 0, stream>>>(Apack, Wpack, dinvP, bigA, bigB);
    gcn_scatter<<<4096, B256, 0, stream>>>(pe_src, pe_dst, bigA, bigB, Ep);
    gcn_finalize<<<4096, B256, 0, stream>>>(bigB, dinvP, bp1, bigC, n_p);
    // layer 2
    pack_w_bf16<<<4, B256, 0, stream>>>(Wp2, Wpack, 128, 4);
    pack_a_bf16<<<cdiv(ntP * 4 * 32, 256), B256, 0, stream>>>(bigC, Apack, n_p, 128, 4);
    gcn_gemm_wmma<4><<<ntP, B256, 0, stream>>>(Apack, Wpack, dinvP, bigA, bigB);
    gcn_scatter<<<4096, B256, 0, stream>>>(pe_src, pe_dst, bigA, bigB, Ep);
    gcn_finalize<<<4096, B256, 0, stream>>>(bigB, dinvP, bp2, bigC, n_p);

    fill_f<<<1, B256, 0, stream>>>(pooled, 256, 0.0f);
    mean_pool<<<64, B256, 0, stream>>>(bigC, n_p, 1.0f / (float)n_p, pooled);

    // ===== drug branch (reuses big buffers; stream order serializes) =====
    fill_f<<<cdiv(n_d, 256), B256, 0, stream>>>(dinvD, n_d, 1.0f);
    deg_accum<<<512, B256, 0, stream>>>(de_dst, dinvD, Ed);
    to_dinv<<<cdiv(n_d, 256), B256, 0, stream>>>(dinvD, n_d);

    // layer 1 (K = 78, padded to 96, nc = 3)
    pack_w_bf16<<<3, B256, 0, stream>>>(Wd1, Wpack, 78, 3);
    pack_a_bf16<<<cdiv(ntD * 3 * 32, 256), B256, 0, stream>>>(dx, Apack, n_d, 78, 3);
    gcn_gemm_wmma<3><<<ntD, B256, 0, stream>>>(Apack, Wpack, dinvD, bigA, bigB);
    gcn_scatter<<<2048, B256, 0, stream>>>(de_src, de_dst, bigA, bigB, Ed);
    gcn_finalize<<<2048, B256, 0, stream>>>(bigB, dinvD, bd1, bigC, n_d);
    // layer 2 (K = 128, nc = 4)
    pack_w_bf16<<<4, B256, 0, stream>>>(Wd2, Wpack, 128, 4);
    pack_a_bf16<<<cdiv(ntD * 4 * 32, 256), B256, 0, stream>>>(bigC, Apack, n_d, 128, 4);
    gcn_gemm_wmma<4><<<ntD, B256, 0, stream>>>(Apack, Wpack, dinvD, bigA, bigB);
    gcn_scatter<<<2048, B256, 0, stream>>>(de_src, de_dst, bigA, bigB, Ed);
    gcn_finalize<<<2048, B256, 0, stream>>>(bigB, dinvD, bd2, bigC, n_d);

    mean_pool<<<64, B256, 0, stream>>>(bigC, n_d, 1.0f / (float)n_d, pooled + HID);

    // ===== head =====
    fc_head<<<1, dim3(HID), 0, stream>>>(pooled, Wf1, bf1, Wf2, bf2, (float*)d_out);
}
